// TransRNN_71244917506425
// MI455X (gfx1250) — matrix operand order repaired and checked by reading, not compiled, not run on hardware
//
#include <hip/hip_runtime.h>
#include <hip/hip_bf16.h>
#include <stdint.h>

typedef __attribute__((ext_vector_type(16))) __bf16 v16bf;
typedef __attribute__((ext_vector_type(8)))  float  v8f;
typedef int v4i __attribute__((vector_size(16)));
typedef __attribute__((address_space(1))) v4i* gas_v4i;
typedef __attribute__((address_space(3))) v4i* las_v4i;

constexpr int kH   = 1024;
constexpr int kL   = 8192;
constexpr int kG3  = 3072;     // 3*H
constexpr int kNB  = 64;       // persistent blocks for the recurrence
constexpr int kUPB = kH / kNB; // hidden units per block = 16

constexpr int kLDA = 40;       // padded LDS strides (x2B = 80B, 16B-aligned)
constexpr int kLDB = 40;

// -------- CDNA5 async global->LDS (ASYNCcnt path), guarded fallback --------
#if defined(__gfx1250__) && __has_builtin(__builtin_amdgcn_global_load_async_to_lds_b128)
#define HAS_ASYNC_LDS 1
#else
#define HAS_ASYNC_LDS 0
#endif

__device__ inline void wait_async0() {
#if HAS_ASYNC_LDS
#if __has_builtin(__builtin_amdgcn_s_wait_asynccnt)
  __builtin_amdgcn_s_wait_asynccnt(0);
#else
  asm volatile("s_wait_asynccnt 0" ::: "memory");
#endif
#endif
}

// ---------------- init: barrier state + h buffers ----------------
__global__ void k_init(unsigned* bar, float* h0, float* h1) {
  int i = threadIdx.x;
  if (i < 2) bar[i] = 0u;
  if (i < kH) { h0[i] = 0.f; h1[i] = 0.f; }
}

// ---------------- W (RxC f32) -> Wt (CxR bf16) (transpose+convert) ----------
__global__ void k_wt(const float* __restrict__ W, __bf16* __restrict__ Wt,
                     int R, int C) {
  int t = blockIdx.x * blockDim.x + threadIdx.x;
  if (t >= R * C) return;
  int r = t / C, c = t % C;
  Wt[(size_t)c * R + r] = (__bf16)W[t];
}

// ---------------- embedding gather -> bf16 ----------------
__global__ void k_gather(const int* __restrict__ src,
                         const float* __restrict__ emb,
                         __bf16* __restrict__ X) {
  int t = blockIdx.x * blockDim.x + threadIdx.x; // kL*kH threads
  int row = t >> 10;
  int k   = t & (kH - 1);
  int idx = src[row];
  X[t] = (__bf16)emb[(size_t)idx * kH + k];
}

// ---------------- f32 -> bf16 convert ----------------
__global__ void k_f2b(const float* __restrict__ A, __bf16* __restrict__ B, int n) {
  int t = blockIdx.x * blockDim.x + threadIdx.x;
  if (t < n) B[t] = (__bf16)A[t];
}

// ---------------- WMMA bf16 GEMM: C(MxN) = A(MxK)*B(KxN) + bias ------------
// Block tile 128x128, BK=32, 256 threads = 8 waves (4 in M x 2 in N).
// Each wave: 2 M-subtiles x 4 N-subtiles -> 8 v_wmma_f32_16x16x32_bf16 / step.
// A tile: async global->LDS B128 copies (ASYNCcnt). B tile: transpose-on-store
// so every fragment read is a ds_load_b128 pair.
// OUTBF=false path streams C with non-temporal stores (Gi is read-once).
template <bool RELU, bool OUTBF>
__global__ __launch_bounds__(256) void k_gemm(
    const __bf16* __restrict__ A, const __bf16* __restrict__ B,
    const float* __restrict__ bias, void* __restrict__ Cout,
    int M, int N, int K) {
  __shared__ __attribute__((aligned(16))) __bf16 sA[128 * kLDA];  // [row][k]
  __shared__ __attribute__((aligned(16))) __bf16 sBt[128 * kLDB]; // [col][k]

  const int tid   = threadIdx.x;
  const int lane  = tid & 31;
  const int wave  = tid >> 5;
  const int hf    = lane >> 4;        // half-wave select (ISA layout)
  const int lrow  = lane & 15;
  const int m0    = blockIdx.y * 128;
  const int n0    = blockIdx.x * 128;
  const int wm    = (wave & 3) * 32;  // wave M offset (2 subtiles of 16)
  const int wn    = (wave >> 2) * 64; // wave N offset (4 subtiles of 16)

  union F { uint4 q[2]; v16bf v; };
  union G { uint4 q[2]; __bf16 e[16]; };

  v8f acc[2][4];
#pragma unroll
  for (int mt = 0; mt < 2; ++mt)
#pragma unroll
    for (int nt = 0; nt < 4; ++nt) acc[mt][nt] = (v8f){};

  for (int kb = 0; kb < K; kb += 32) {
    { // A tile 128x32 -> sA row-major: thread = 16 contig elems of one row
      int row = tid >> 1;
      int seg = (tid & 1) * 16;
      const __bf16* gp = A + (size_t)(m0 + row) * K + kb + seg;
      __bf16* lp = &sA[row * kLDA + seg];
#if HAS_ASYNC_LDS
      // INST_OFFSET is added to both global and LDS addresses (ISA 10.x):
      // two B128 async copies move this thread's 32B without touching VGPRs.
      __builtin_amdgcn_global_load_async_to_lds_b128((gas_v4i)gp, (las_v4i)lp,
                                                     0, 0);
      __builtin_amdgcn_global_load_async_to_lds_b128((gas_v4i)gp, (las_v4i)lp,
                                                     16, 0);
#else
      const uint4* ga = (const uint4*)gp;
      *(uint4*)lp       = ga[0];
      *(uint4*)(lp + 8) = ga[1];
#endif
    }
    { // B tile 32x128 -> sBt transposed ([col][k])
      int row  = tid >> 3;          // k within tile: 0..31
      int segc = (tid & 7) * 16;    // col base: 0..112
      G g;
      const uint4* gb = (const uint4*)(B + (size_t)(kb + row) * N + n0 + segc);
      g.q[0] = gb[0];
      g.q[1] = gb[1];
#pragma unroll
      for (int j = 0; j < 16; ++j) sBt[(segc + j) * kLDB + row] = g.e[j];
    }
    wait_async0();
    __syncthreads();

    // Fragments per CDNA5 16-bit WMMA layouts (ISA 7.12.2): per lane,
    // A = rows wm+mt*16+lrow, K runs [8hf..8hf+7] and [16+8hf..23+8hf];
    // B = col wn+nt*16+lrow, K runs [8hf..8hf+7] and [16+8hf..23+8hf].
    F a[2], b[4];
#pragma unroll
    for (int mt = 0; mt < 2; ++mt) {
      const __bf16* pa = sA + (wm + mt * 16 + lrow) * kLDA + hf * 8;
      a[mt].q[0] = *(const uint4*)pa;
      a[mt].q[1] = *(const uint4*)(pa + 16);
    }
#pragma unroll
    for (int nt = 0; nt < 4; ++nt) {
      const __bf16* pb = sBt + (wn + nt * 16 + lrow) * kLDB + hf * 8;
      b[nt].q[0] = *(const uint4*)pb;
      b[nt].q[1] = *(const uint4*)(pb + 16);
    }
#pragma unroll
    for (int mt = 0; mt < 2; ++mt)
#pragma unroll
      for (int nt = 0; nt < 4; ++nt)
        acc[mt][nt] = __builtin_amdgcn_wmma_f32_16x16x32_bf16(
            false, a[mt].v, false, b[nt].v, (short)0, acc[mt][nt], false, false);
    __syncthreads();
  }

  // Epilogue (straight-line, compile-time variant). C 16x16 f32 layout:
  // VGPR r -> M = r (+8 for upper half-wave), N = lane%16.
#pragma unroll
  for (int mt = 0; mt < 2; ++mt) {
#pragma unroll
    for (int nt = 0; nt < 4; ++nt) {
      int nn = n0 + wn + nt * 16 + lrow;
      float bv = bias[nn];
#pragma unroll
      for (int r = 0; r < 8; ++r) {
        int mm = m0 + wm + mt * 16 + r + hf * 8;
        float v = acc[mt][nt][r] + bv;
        if (RELU) v = fmaxf(v, 0.f);
        if (OUTBF) {
          ((__bf16*)Cout)[(size_t)mm * N + nn] = (__bf16)v;
        } else {
          // Gi is a 96MB read-once stream: keep it out of L2 (TH=NT) so the
          // recurrent weights stay resident.
          __builtin_nontemporal_store(v, (float*)Cout + (size_t)mm * N + nn);
        }
      }
    }
  }
}

// ---------------- device-wide split barrier ----------------
__device__ inline void grid_barrier(unsigned* cnt, unsigned* gen) {
  __threadfence();
  __syncthreads();
  if (threadIdx.x == 0) {
    unsigned g = __atomic_load_n(gen, __ATOMIC_RELAXED);
    unsigned arrived = atomicAdd(cnt, 1u);
    if (arrived == (unsigned)(kNB - 1)) {
      __atomic_store_n(cnt, 0u, __ATOMIC_RELAXED);
      __threadfence();
      atomicAdd(gen, 1u);
    } else {
      while (__atomic_load_n(gen, __ATOMIC_RELAXED) == g)
        __builtin_amdgcn_s_sleep(2);
    }
    __threadfence();
  }
  __syncthreads();
}

// ---------------- persistent GRU scan (sequential part) ----------------
__global__ __launch_bounds__(256) void k_gru(
    const float* __restrict__ Whh, const float* __restrict__ bhh,
    const float* __restrict__ Gi, float* hbuf0, float* hbuf1,
    float* __restrict__ out, unsigned* bar, int write_out) {
  __shared__ float sh[kH];
  __shared__ float sg[3 * kUPB];
  const int tid  = threadIdx.x;
  const int lane = tid & 31;
  const int wave = tid >> 5;
  const int u0   = blockIdx.x * kUPB;

  for (int t = 0; t < kL; ++t) {
    const float* hc = (t & 1) ? hbuf1 : hbuf0;
    float*       hn = (t & 1) ? hbuf0 : hbuf1;

    for (int i = tid; i < kH; i += 256)
      sh[i] = ((const volatile float*)hc)[i];
    if (tid == 0 && t + 1 < kL)
      __builtin_prefetch(Gi + (size_t)(t + 1) * kG3 + u0, 0, 0);
    __syncthreads();

#pragma unroll
    for (int rr = 0; rr < 6; ++rr) {
      int lr   = wave * 6 + rr;          // 0..47
      int gate = lr >> 4;
      int unit = lr & 15;
      const float* wrow = Whh + (size_t)(gate * kH + u0 + unit) * kH;
      float acc = 0.f;
#pragma unroll 4
      for (int k = lane; k < kH; k += 32) acc += wrow[k] * sh[k];
      for (int off = 16; off > 0; off >>= 1) acc += __shfl_xor(acc, off, 32);
      if (lane == 0) sg[lr] = acc;
    }
    __syncthreads();

    if (tid < kUPB) {
      int u = u0 + tid;
      const float* gi = Gi + (size_t)t * kG3;   // b_ih already folded in GEMM
      float g_r = __builtin_nontemporal_load(gi + u);
      float g_z = __builtin_nontemporal_load(gi + kH + u);
      float g_n = __builtin_nontemporal_load(gi + 2 * kH + u);
      float r = 1.f / (1.f + __expf(-(g_r + sg[tid]            + bhh[u])));
      float z = 1.f / (1.f + __expf(-(g_z + sg[kUPB + tid]     + bhh[kH + u])));
      float n = tanhf(g_n + r * (sg[2 * kUPB + tid] + bhh[2 * kH + u]));
      float hv = (1.f - z) * n + z * sh[u];
      hn[u] = hv;
      if (write_out) out[(size_t)t * kH + u] = hv;
    }
    grid_barrier(bar, bar + 1);
  }
}

// ---------------- host ----------------
extern "C" void kernel_launch(void* const* d_in, const int* in_sizes, int n_in,
                              void* d_out, int out_size, void* d_ws, size_t ws_size,
                              hipStream_t stream) {
  const int*   src  = (const int*)  d_in[0];
  const float* tgt  = (const float*)d_in[1];
  const float* emb  = (const float*)d_in[2];
  const float* eWih = (const float*)d_in[3];
  const float* eWhh = (const float*)d_in[4];
  const float* eBih = (const float*)d_in[5];
  const float* eBhh = (const float*)d_in[6];
  const float* dEw  = (const float*)d_in[7];
  const float* dEb  = (const float*)d_in[8];
  const float* dWih = (const float*)d_in[9];
  const float* dWhh = (const float*)d_in[10];
  const float* dBih = (const float*)d_in[11];
  const float* dBhh = (const float*)d_in[12];
  float* out = (float*)d_out;

  char* ws = (char*)d_ws;
  unsigned* bar = (unsigned*)ws;                       // 2 x u32
  float* h0 = (float*)(ws + 256);
  float* h1 = h0 + kH;
  __bf16* Wt = (__bf16*)(ws + 16384);                  // 1024x3072 bf16 (6 MB)
  __bf16* Xb = (__bf16*)(ws + 16384 + 6291456);        // 8192x1024 bf16 (16 MB)
  __bf16* Yb = Xb + (size_t)kL * kH;                   // 8192x1024 bf16 (16 MB)
  float*  Gi = (float*)(ws + 16384 + 6291456 + 2 * (size_t)kL * kH * 2); // 96 MB

  k_init<<<1, 1024, 0, stream>>>(bar, h0, h1);

  // ----- encoder: Gi = embed(src) @ W_ih^T + b_ih, then scan -----
  k_wt<<<(kG3 * kH) / 256, 256, 0, stream>>>(eWih, Wt, kG3, kH);
  k_gather<<<(kL * kH) / 256, 256, 0, stream>>>(src, emb, Xb);
  k_gemm<false, false><<<dim3(kG3 / 128, kL / 128), 256, 0, stream>>>(
      Xb, Wt, eBih, Gi, kL, kG3, kH);
  k_gru<<<kNB, 256, 0, stream>>>(eWhh, eBhh, Gi, h0, h1, out, bar, 0);

  // ----- decoder input projection: Yb = relu(tgt @ dec_emb_W^T + b) (bf16) --
  k_wt<<<(kH * kH) / 256, 256, 0, stream>>>(dEw, Wt, kH, kH);
  k_f2b<<<(kL * kH) / 256, 256, 0, stream>>>(tgt, Xb, kL * kH);
  k_gemm<true, true><<<dim3(kH / 128, kL / 128), 256, 0, stream>>>(
      Xb, Wt, dEb, Yb, kL, kH, kH);

  // ----- decoder: Gi = Yb @ dec_W_ih^T + b_ih, then scan (writes d_out) -----
  k_wt<<<(kG3 * kH) / 256, 256, 0, stream>>>(dWih, Wt, kG3, kH);
  k_gemm<false, false><<<dim3(kG3 / 128, kL / 128), 256, 0, stream>>>(
      Yb, Wt, dBih, Gi, kL, kG3, kH);
  k_gru<<<kNB, 256, 0, stream>>>(dWhh, dBhh, Gi, h0, h1, out, bar, 1);
}
